// AstraloraLayer_31903017075277
// MI455X (gfx1250) — compile-verified, matrix operand order checked by reading, not run on hardware
//
#include <hip/hip_runtime.h>

// ---------------------------------------------------------------------------
// Y[M, D_OUT] = X[M, D_INP] @ W[D_OUT, D_INP]^T    (forward value == y_bb)
// Strategy (MI455X / gfx1250, wave32):
//   Pass 1: split fp32 -> bf16 hi + bf16 lo residual (planar, in d_ws).
//   Pass 2: plain bf16 WMMA GEMM over 3 K-phases (hi*hi + lo*hi + hi*lo),
//           fp32 accumulate -> ~2^-16 relative error at 3x bf16-GEMM cost.
//   GEMM tile: 256x128 block, 8 waves, 64x64 per wave (4x4 WMMA tiles)
//   -> 16 ds_load_b128 feed 16 v_wmma per K-step (1:1), XDL-bound.
// Fallback (small ws): fused in-loop split kernel.
// ---------------------------------------------------------------------------

typedef __attribute__((ext_vector_type(16))) __bf16 v16bf;
typedef __attribute__((ext_vector_type(8)))  __bf16 v8bf;
typedef __attribute__((ext_vector_type(4)))  __bf16 v4bf;
typedef __attribute__((ext_vector_type(8)))  float  v8f;
typedef __attribute__((ext_vector_type(4)))  float  v4f;

#define D_INP 4096
#define D_OUT 4096

#define BM 256          // block tile M
#define BN 128          // block tile N
#define BK 32           // K step (one bf16 WMMA K)
#define LDSS 40         // padded LDS row stride (80B rows -> conflict-free b128)

__device__ __forceinline__ __bf16 f2bf(float f) { return (__bf16)f; }

// ---------------------------------------------------------------------------
// Pass 1: planar fp32 -> bf16 hi/lo split. Memory bound.
// ---------------------------------------------------------------------------
__global__ __launch_bounds__(256)
void split_f32_bf16x2(const float* __restrict__ src,
                      __bf16* __restrict__ hi,
                      __bf16* __restrict__ lo)
{
    const size_t i = ((size_t)blockIdx.x * 256 + threadIdx.x) * 4;
    v4f v = *(const v4f*)(src + i);
    v4bf h, l;
#pragma unroll
    for (int e = 0; e < 4; ++e) {
        __bf16 hb = f2bf(v[e]);
        h[e] = hb;
        l[e] = f2bf(v[e] - (float)hb);
    }
    *(v4bf*)(hi + i) = h;
    *(v4bf*)(lo + i) = l;
}

// ---------------------------------------------------------------------------
// Pass 2: plain bf16 GEMM, 3 K-phases over planar hi/lo operands.
// 8 wave32s arranged 4(M) x 2(N); each wave owns 64x64 = 4x4 WMMA tiles.
// ---------------------------------------------------------------------------
__global__ __launch_bounds__(256)
void astralora_gemm_bf16_planar(const __bf16* __restrict__ Xhi,
                                const __bf16* __restrict__ Xlo,
                                const __bf16* __restrict__ Whi,
                                const __bf16* __restrict__ Wlo,
                                float* __restrict__ Y)
{
    __shared__ __align__(16) __bf16 sX[BM * LDSS];   // 20.0 KB
    __shared__ __align__(16) __bf16 sW[BN * LDSS];   // 10.0 KB

    const int tid  = threadIdx.x;
    const int bn   = blockIdx.x;
    const int bm   = blockIdx.y;
    const int wave = tid >> 5;
    const int lane = tid & 31;

    const int wm = (wave & 3) << 6;       // wave M origin within block (0..192)
    const int wn = (wave >> 2) << 6;      // wave N origin within block (0 or 64)

    // X loader: 1 thread per row, 32 bf16 (4x b128). W loader: 2 threads/row.
    const int sox = tid * LDSS;
    const int sow = (tid >> 1) * LDSS + ((tid & 1) << 4);

    const size_t xoff = (size_t)(bm * BM + tid) * D_INP;
    const size_t woff = (size_t)(bn * BN + (tid >> 1)) * D_INP + ((tid & 1) << 4);

    // fragment addressing: lane L -> row (L%16), K halves (L/16)*8 and +16
    const int frow = lane & 15;
    const int fk   = (lane >> 4) << 3;

    v8f acc[4][4];
#pragma unroll
    for (int i = 0; i < 4; ++i)
#pragma unroll
        for (int j = 0; j < 4; ++j)
            acc[i][j] = (v8f){0.f, 0.f, 0.f, 0.f, 0.f, 0.f, 0.f, 0.f};

    // three phases: (Xhi,Whi), (Xlo,Whi), (Xhi,Wlo) -- shared accumulators
#pragma unroll 1
    for (int ph = 0; ph < 3; ++ph) {
        const __bf16* xg = ((ph == 1) ? Xlo : Xhi) + xoff;
        const __bf16* wg = ((ph == 2) ? Wlo : Whi) + woff;

        // register prefetch of K-step 0
        v8bf gx[4], gw[2];
#pragma unroll
        for (int q = 0; q < 4; ++q) gx[q] = *(const v8bf*)(xg + (q << 3));
        gw[0] = *(const v8bf*)(wg);
        gw[1] = *(const v8bf*)(wg + 8);

#pragma unroll 1
        for (int k = 0; k < D_INP; k += BK) {
            __syncthreads();                      // prev iter's LDS reads done
#pragma unroll
            for (int q = 0; q < 4; ++q) *(v8bf*)&sX[sox + (q << 3)] = gx[q];
            *(v8bf*)&sW[sow]     = gw[0];
            *(v8bf*)&sW[sow + 8] = gw[1];
            __syncthreads();                      // tile visible

            const int kn = k + BK;                // prefetch next K-step
            if (kn < D_INP) {
#pragma unroll
                for (int q = 0; q < 4; ++q) gx[q] = *(const v8bf*)(xg + kn + (q << 3));
                gw[0] = *(const v8bf*)(wg + kn);
                gw[1] = *(const v8bf*)(wg + kn + 8);
            }

            v16bf a[4], b[4];
#pragma unroll
            for (int i = 0; i < 4; ++i) {
                const int ro = (wm + (i << 4) + frow) * LDSS + fk;
                v8bf h0 = *(const v8bf*)&sX[ro];
                v8bf h1 = *(const v8bf*)&sX[ro + 16];
#pragma unroll
                for (int e = 0; e < 8; ++e) { a[i][e] = h0[e]; a[i][e + 8] = h1[e]; }
            }
#pragma unroll
            for (int j = 0; j < 4; ++j) {
                const int ro = (wn + (j << 4) + frow) * LDSS + fk;
                v8bf h0 = *(const v8bf*)&sW[ro];
                v8bf h1 = *(const v8bf*)&sW[ro + 16];
#pragma unroll
                for (int e = 0; e < 8; ++e) { b[j][e] = h0[e]; b[j][e + 8] = h1[e]; }
            }

#pragma unroll
            for (int i = 0; i < 4; ++i)
#pragma unroll
                for (int j = 0; j < 4; ++j)
                    acc[i][j] = __builtin_amdgcn_wmma_f32_16x16x32_bf16(
                        false, a[i], false, b[j], (short)0, acc[i][j], false, false);
        }
    }

    // epilogue: VGPR r of lane L holds (M = (L/16)*8 + r, N = L%16)
    const size_t crow0 = (size_t)bm * BM + wm + ((lane >> 4) << 3);
    const size_t ccol  = (size_t)bn * BN + wn + (lane & 15);
#pragma unroll
    for (int i = 0; i < 4; ++i)
#pragma unroll
        for (int j = 0; j < 4; ++j)
#pragma unroll
            for (int r = 0; r < 8; ++r)
                Y[(crow0 + (size_t)(i << 4) + r) * D_OUT + ccol + (j << 4)] = acc[i][j][r];
}

// ---------------------------------------------------------------------------
// Fallback: fused in-loop split (128x128 block), used when ws is too small.
// ---------------------------------------------------------------------------
#define FBM 128

__global__ __launch_bounds__(256)
void astralora_gemm_bf16x3(const float* __restrict__ X,
                           const float* __restrict__ W,
                           float* __restrict__ Y)
{
    __shared__ __align__(16) __bf16 sXh[FBM * LDSS];
    __shared__ __align__(16) __bf16 sXl[FBM * LDSS];
    __shared__ __align__(16) __bf16 sWh[BN * LDSS];
    __shared__ __align__(16) __bf16 sWl[BN * LDSS];

    const int tid  = threadIdx.x;
    const int bn   = blockIdx.x;
    const int bm   = blockIdx.y;
    const int wave = tid >> 5;
    const int lane = tid & 31;
    const int wm = (wave & 3) << 5;
    const int wn = (wave >> 2) << 6;
    const int lrow = tid >> 1;
    const int lcol = (tid & 1) << 4;

    const float* xg = X + (size_t)(bm * FBM + lrow) * D_INP + lcol;
    const float* wg = W + (size_t)(bn * BN + lrow) * D_INP + lcol;

    v8f acc[2][4];
#pragma unroll
    for (int i = 0; i < 2; ++i)
#pragma unroll
        for (int j = 0; j < 4; ++j)
            acc[i][j] = (v8f){0.f, 0.f, 0.f, 0.f, 0.f, 0.f, 0.f, 0.f};

    const int frow = lane & 15;
    const int fk   = (lane >> 4) << 3;

    for (int k = 0; k < D_INP; k += BK) {
        v4f xv[4], wv[4];
        const v4f* xp = (const v4f*)(xg + k);
        const v4f* wp = (const v4f*)(wg + k);
#pragma unroll
        for (int i = 0; i < 4; ++i) { xv[i] = xp[i]; wv[i] = wp[i]; }

        v8bf xh[2], xl[2], wh[2], wl[2];
#pragma unroll
        for (int i = 0; i < 16; ++i) {
            float fx = xv[i >> 2][i & 3];
            __bf16 hx = f2bf(fx);
            xh[i >> 3][i & 7] = hx;
            xl[i >> 3][i & 7] = f2bf(fx - (float)hx);
            float fw = wv[i >> 2][i & 3];
            __bf16 hw = f2bf(fw);
            wh[i >> 3][i & 7] = hw;
            wl[i >> 3][i & 7] = f2bf(fw - (float)hw);
        }

        __syncthreads();
        {
            const int so = lrow * LDSS + lcol;
            *(v8bf*)&sXh[so]     = xh[0];
            *(v8bf*)&sXh[so + 8] = xh[1];
            *(v8bf*)&sXl[so]     = xl[0];
            *(v8bf*)&sXl[so + 8] = xl[1];
            *(v8bf*)&sWh[so]     = wh[0];
            *(v8bf*)&sWh[so + 8] = wh[1];
            *(v8bf*)&sWl[so]     = wl[0];
            *(v8bf*)&sWl[so + 8] = wl[1];
        }
        __syncthreads();

        v16bf a_h[2], a_l[2], b_h[4], b_l[4];
#pragma unroll
        for (int i = 0; i < 2; ++i) {
            const int ro = (wm + (i << 4) + frow) * LDSS + fk;
            v8bf h0 = *(const v8bf*)&sXh[ro];
            v8bf h1 = *(const v8bf*)&sXh[ro + 16];
            v8bf l0 = *(const v8bf*)&sXl[ro];
            v8bf l1 = *(const v8bf*)&sXl[ro + 16];
#pragma unroll
            for (int e = 0; e < 8; ++e) {
                a_h[i][e] = h0[e]; a_h[i][e + 8] = h1[e];
                a_l[i][e] = l0[e]; a_l[i][e + 8] = l1[e];
            }
        }
#pragma unroll
        for (int j = 0; j < 4; ++j) {
            const int ro = (wn + (j << 4) + frow) * LDSS + fk;
            v8bf h0 = *(const v8bf*)&sWh[ro];
            v8bf h1 = *(const v8bf*)&sWh[ro + 16];
            v8bf l0 = *(const v8bf*)&sWl[ro];
            v8bf l1 = *(const v8bf*)&sWl[ro + 16];
#pragma unroll
            for (int e = 0; e < 8; ++e) {
                b_h[j][e] = h0[e]; b_h[j][e + 8] = h1[e];
                b_l[j][e] = l0[e]; b_l[j][e + 8] = l1[e];
            }
        }

#pragma unroll
        for (int i = 0; i < 2; ++i)
#pragma unroll
            for (int j = 0; j < 4; ++j) {
                acc[i][j] = __builtin_amdgcn_wmma_f32_16x16x32_bf16(
                    false, a_h[i], false, b_h[j], (short)0, acc[i][j], false, false);
                acc[i][j] = __builtin_amdgcn_wmma_f32_16x16x32_bf16(
                    false, a_h[i], false, b_l[j], (short)0, acc[i][j], false, false);
                acc[i][j] = __builtin_amdgcn_wmma_f32_16x16x32_bf16(
                    false, a_l[i], false, b_h[j], (short)0, acc[i][j], false, false);
            }
    }

    const size_t crow0 = (size_t)bm * FBM + wm + ((lane >> 4) << 3);
    const size_t ccol  = (size_t)bn * BN + wn + (lane & 15);
#pragma unroll
    for (int i = 0; i < 2; ++i)
#pragma unroll
        for (int j = 0; j < 4; ++j)
#pragma unroll
            for (int r = 0; r < 8; ++r)
                Y[(crow0 + (size_t)(i << 4) + r) * D_OUT + ccol + (j << 4)] = acc[i][j][r];
}

// ---------------------------------------------------------------------------
extern "C" void kernel_launch(void* const* d_in, const int* in_sizes, int n_in,
                              void* d_out, int out_size, void* d_ws, size_t ws_size,
                              hipStream_t stream) {
    (void)n_in; (void)out_size;
    const float* x = (const float*)d_in[0];   // [8, 2048, 4096] fp32
    const float* w = (const float*)d_in[1];   // [4096*4096]     fp32
    // d_in[2..4] = U, S, V: gradient-only surrogate; forward value == x @ A^T.
    float* y = (float*)d_out;

    const size_t NX = (size_t)in_sizes[0];    // 67,108,864
    const size_t NW = (size_t)in_sizes[1];    // 16,777,216
    const int Mrows = in_sizes[0] / D_INP;    // 16384

    const size_t need = (2 * NX + 2 * NW) * sizeof(__bf16);  // ~320 MB
    if (ws_size >= need) {
        __bf16* xhi = (__bf16*)d_ws;
        __bf16* xlo = xhi + NX;
        __bf16* whi = xlo + NX;
        __bf16* wlo = whi + NW;
        split_f32_bf16x2<<<dim3((unsigned)(NX / 1024)), dim3(256), 0, stream>>>(x, xhi, xlo);
        split_f32_bf16x2<<<dim3((unsigned)(NW / 1024)), dim3(256), 0, stream>>>(w, whi, wlo);
        dim3 grid(D_OUT / BN, Mrows / BM);    // (32, 64)
        astralora_gemm_bf16_planar<<<grid, dim3(256), 0, stream>>>(xhi, xlo, whi, wlo, y);
    } else {
        dim3 grid(D_OUT / BN, Mrows / FBM);   // (32, 128)
        astralora_gemm_bf16x3<<<grid, dim3(256), 0, stream>>>(x, w, y);
    }
}
